// Step_1_33818572489087
// MI455X (gfx1250) — matrix operand order repaired and checked by reading, not compiled, not run on hardware
//
#include <hip/hip_runtime.h>
#include <hip/hip_bf16.h>
#include <math.h>

// ---------------- problem constants ----------------
constexpr int Bb  = 16;
constexpr int Ss  = 1024;
constexpr int Dd  = 768;
constexpr int NHh = 4;
constexpr int HDd = 192;    // Dd / NHh
constexpr int DFFf= 3072;
constexpr int D3  = 2304;   // 3*Dd

// ---------------- WMMA types ----------------
typedef __bf16 v16bf __attribute__((ext_vector_type(16)));
typedef float  v8f   __attribute__((ext_vector_type(8)));

union FragU { v16bf v; uint4 q[2]; };

__device__ inline v8f vzero() {
  v8f z;
#pragma unroll
  for (int i = 0; i < 8; ++i) z[i] = 0.0f;
  return z;
}

// A-matrix fragment (16 rows x 32 K, bf16).  ISA layout:
// lanes 0-15 : M=lane, elems 0..7 = K k0..k0+7, elems 8..15 = K k0+16..k0+23
// lanes 16-31: M=lane-16, elems 0..7 = K k0+8..k0+15, 8..15 = K k0+24..k0+31
__device__ inline v16bf load_fragA(const __hip_bfloat16* base, int ld, int k0) {
  int lane = threadIdx.x & 31;
  int r = lane & 15, hf = lane >> 4;
  const __hip_bfloat16* p = base + (size_t)r * ld + k0 + hf * 8;
  FragU f;
  f.q[0] = *reinterpret_cast<const uint4*>(p);
  f.q[1] = *reinterpret_cast<const uint4*>(p + 16);
  return f.v;
}

// B-matrix fragment (32 K x 16 N, bf16), sourced from W[N][K] rows.  ISA layout:
// lanes 0-15 : N=lane, elems 0..7 = K k0..k0+7, elems 8..15 = K k0+16..k0+23
// lanes 16-31: N=lane-16, elems 0..7 = K k0+8..k0+15, 8..15 = K k0+24..k0+31
__device__ inline v16bf load_fragB(const __hip_bfloat16* base, int ld, int k0) {
  int lane = threadIdx.x & 31;
  int r = lane & 15, hf = lane >> 4;
  const __hip_bfloat16* p = base + (size_t)r * ld + k0 + hf * 8;
  FragU f;
  f.q[0] = *reinterpret_cast<const uint4*>(p);
  f.q[1] = *reinterpret_cast<const uint4*>(p + 16);
  return f.v;
}

// ---- gfx1250 async global->LDS DMA (ASYNCcnt-tracked; writes LDS directly) ----
// LDS address comes from the vdst VGPR; generic LDS pointers truncate to the LDS offset
// in their low 32 bits (ISA 10.2), so we pass that as the LDS-address VGPR.
__device__ inline void async_ld_b128(unsigned lds_off, const void* gaddr) {
  asm volatile("global_load_async_to_lds_b128 %0, %1, off"
               :: "v"(lds_off), "v"(gaddr) : "memory");
}
__device__ inline void wait_async0() {
  asm volatile("s_wait_asynccnt 0" ::: "memory");
}

__device__ inline float gelu_exact(float x) {
  return 0.5f * x * (1.0f + erff(x * 0.70710678118654752f));
}

// ---------------- f32 -> bf16 convert ----------------
__global__ void __launch_bounds__(256) cvt_f32_bf16(const float* __restrict__ in,
                                                    __hip_bfloat16* __restrict__ out, int n) {
  int i = blockIdx.x * 256 + threadIdx.x;
  if (i < n) out[i] = __float2bfloat16(in[i]);
}

// ---------------- generic WMMA GEMM:  C[M,N] = act(A[M,K] @ W[N,K]^T + bias) ----------------
// grid = (N/128, ceil(M/128)), block = 256 (8 waves); each wave -> 16 x 128 C tile.
// LDS double-buffered staging via GLOBAL_LOAD_ASYNC_TO_LDS_B128: tile kt+1 DMAs into the
// spare LDS buffer while the 8 WMMAs of tile kt run from ds_load_b128 fragments; a single
// s_wait_asynccnt 0 + barrier publishes it.  No VGPR round-trip, no s_wait_loadcnt on the
// WMMA path.  B tile fetched once per block.  LDS rows padded 32 -> 40 elems (bank spread);
// 2*2*128*40*2 B = 40 KB of the 320 KB WGP pool.
__global__ void __launch_bounds__(256)
gemm_bf16_wmma(const __hip_bfloat16* __restrict__ A, int lda,
               const __hip_bfloat16* __restrict__ W, int ldw,
               const float* __restrict__ bias,
               float* __restrict__ Cf, __hip_bfloat16* __restrict__ Cb, int ldc,
               int M, int K, int act) {
  __shared__ __hip_bfloat16 sA[2][128][40];
  __shared__ __hip_bfloat16 sB[2][128][40];
  constexpr unsigned BUFBYTES = 128 * 40 * 2;   // 10240 B per buffer

  int tid = threadIdx.x;
  int wave = tid >> 5;
  int lane = tid & 31;
  int m0 = blockIdx.y * 128 + wave * 16;
  int n0 = blockIdx.x * 128;
  int mcap = (M > 16) ? (M - 16) : 0;
  int m0a = (m0 < M) ? m0 : mcap;          // clamp (small-M case); clamped waves skip stores

  // cooperative staging coords: thread -> (row 0..127, 16-element half)
  int crow = tid >> 1;
  int csub = (tid & 1) * 16;
  int arow = blockIdx.y * 128 + crow; if (arow > M - 1) arow = M - 1;
  const __hip_bfloat16* gA = A + (size_t)arow * lda + csub;
  const __hip_bfloat16* gB = W + (size_t)(n0 + crow) * ldw + csub;
  unsigned ldsA = (unsigned)(size_t)&sA[0][crow][csub];
  unsigned ldsB = (unsigned)(size_t)&sB[0][crow][csub];

  // ---- prologue: async-stage tile 0 ----
  async_ld_b128(ldsA,      gA);
  async_ld_b128(ldsA + 16, gA + 8);
  async_ld_b128(ldsB,      gB);
  async_ld_b128(ldsB + 16, gB + 8);
  wait_async0();
  __syncthreads();

  v8f acc[8];
#pragma unroll
  for (int j = 0; j < 8; ++j) acc[j] = vzero();

  int nk = K >> 5;
  for (int kt = 0; kt < nk; ++kt) {
    int buf = kt & 1;
    bool more = (kt + 1) < nk;
    if (more) {
      int k1 = (kt + 1) * 32;
      unsigned nb = (unsigned)(buf ^ 1) * BUFBYTES;
      async_ld_b128(ldsA + nb,      gA + k1);
      async_ld_b128(ldsA + nb + 16, gA + k1 + 8);
      async_ld_b128(ldsB + nb,      gB + k1);
      async_ld_b128(ldsB + nb + 16, gB + k1 + 8);
      if (kt + 2 < nk) {
        __builtin_prefetch(gA + k1 + 32, 0, 3);
        __builtin_prefetch(gB + k1 + 32, 0, 3);
      }
    }
    // ---- compute phase out of LDS (low-latency ds_load fragments) ----
    v16bf a = load_fragA(&sA[buf][wave * 16][0], 40, 0);
#pragma unroll
    for (int j = 0; j < 8; ++j) {
      v16bf bfr = load_fragB(&sB[buf][j * 16][0], 40, 0);
      acc[j] = __builtin_amdgcn_wmma_f32_16x16x32_bf16(
          false, a, false, bfr, (short)0, acc[j], false, false);
    }
    if (more) wait_async0();   // tile kt+1 has landed in LDS
    __syncthreads();
  }

  if (m0 != m0a) return;                    // duplicated (clamped) wave: no store
  int hf = lane >> 4, nl = lane & 15;
#pragma unroll
  for (int j = 0; j < 8; ++j) {
    int n = n0 + j * 16 + nl;
    float bv = bias ? bias[n] : 0.0f;
#pragma unroll
    for (int i = 0; i < 8; ++i) {
      int row = m0 + i + 8 * hf;
      float v = acc[j][i] + bv;
      if (act == 1) v = gelu_exact(v);
      if (Cf) Cf[(size_t)row * ldc + n] = v;
      if (Cb) Cb[(size_t)row * ldc + n] = __float2bfloat16(v);
    }
  }
}

// ---------------- attention: one wave per (b, h, 16-row q tile) ----------------
// dynamic LDS: 16*1024 f32 scores (64KB) + 16*1024 bf16 P (32KB) = 96KB (320KB WGP pool)
__global__ void __launch_bounds__(32)
attn_wmma(const __hip_bfloat16* __restrict__ QKV,
          const __hip_bfloat16* __restrict__ Vt,
          const int* __restrict__ mask,
          __hip_bfloat16* __restrict__ ctx) {
  extern __shared__ char smem[];
  float* sS = reinterpret_cast<float*>(smem);                          // [16][S]
  __hip_bfloat16* sP = reinterpret_cast<__hip_bfloat16*>(smem + 16 * Ss * 4);  // [16][S]

  int t = blockIdx.x;
  int qt = t & 63;            // S/16 = 64
  int h  = (t >> 6) & 3;
  int b  = t >> 8;
  int q0 = qt * 16;
  int lane = threadIdx.x;
  int hf = lane >> 4, nl = lane & 15;

  const __hip_bfloat16* Qb = QKV + (size_t)(b * Ss + q0) * D3 + h * HDd;
  const float scale = 0.07216878364870323f;   // 1/sqrt(192)

  // Q fragments are invariant across all key blocks: hoist (6 frags = 48 VGPRs).
  v16bf qf[6];
#pragma unroll
  for (int i = 0; i < 6; ++i) qf[i] = load_fragA(Qb, D3, i * 32);

  // ---- phase 1: scores = mask(QK^T * scale) into LDS ----
  for (int kb = 0; kb < Ss; kb += 16) {
    const __hip_bfloat16* Kb = QKV + (size_t)(b * Ss + kb) * D3 + Dd + h * HDd;
    v16bf kf[6];
#pragma unroll
    for (int i = 0; i < 6; ++i) kf[i] = load_fragB(Kb, D3, i * 32);
    v8f acc = vzero();
#pragma unroll
    for (int i = 0; i < 6; ++i)
      acc = __builtin_amdgcn_wmma_f32_16x16x32_bf16(
          false, qf[i], false, kf[i], (short)0, acc, false, false);
    int key = kb + nl;
    bool dead = (mask[b * Ss + key] == 0);
#pragma unroll
    for (int i = 0; i < 8; ++i) {
      float v = dead ? -1e30f : acc[i] * scale;
      sS[(i + 8 * hf) * Ss + key] = v;
    }
  }
  __syncthreads();

  // ---- phase 2: softmax; 2 lanes per row (halves combined via shfl_xor 16) ----
  {
    int r = lane & 15;
    int c0 = (lane >> 4) * (Ss / 2);
    float* row = sS + r * Ss;
    float mx = -3.4e38f;
    for (int j = 0; j < Ss / 2; ++j) mx = fmaxf(mx, row[c0 + j]);
    mx = fmaxf(mx, __shfl_xor(mx, 16, 32));
    float sum = 0.0f;
    for (int j = 0; j < Ss / 2; ++j) {
      float e = expf(row[c0 + j] - mx);
      row[c0 + j] = e;
      sum += e;
    }
    sum += __shfl_xor(sum, 16, 32);
    float inv = 1.0f / sum;
    __hip_bfloat16* prow = sP + r * Ss;
    for (int j = 0; j < Ss / 2; ++j) prow[c0 + j] = __float2bfloat16(row[c0 + j] * inv);
  }
  __syncthreads();

  // ---- phase 3: ctx = P @ V  (A = P from LDS, B = Vt rows, contiguous) ----
  v8f acc[12];
#pragma unroll
  for (int j = 0; j < 12; ++j) acc[j] = vzero();
  const __hip_bfloat16* Vb = Vt + (size_t)((b * NHh + h) * HDd) * Ss;
  for (int k0 = 0; k0 < Ss; k0 += 32) {
    v16bf a = load_fragA(sP, Ss, k0);
    v16bf bf[6];
#pragma unroll
    for (int j = 0; j < 6; ++j) bf[j] = load_fragB(Vb + (size_t)(j * 16) * Ss, Ss, k0);
#pragma unroll
    for (int j = 0; j < 6; ++j)
      acc[j] = __builtin_amdgcn_wmma_f32_16x16x32_bf16(
          false, a, false, bf[j], (short)0, acc[j], false, false);
#pragma unroll
    for (int j = 0; j < 6; ++j) bf[j] = load_fragB(Vb + (size_t)((j + 6) * 16) * Ss, Ss, k0);
#pragma unroll
    for (int j = 0; j < 6; ++j)
      acc[j + 6] = __builtin_amdgcn_wmma_f32_16x16x32_bf16(
          false, a, false, bf[j], (short)0, acc[j + 6], false, false);
  }
#pragma unroll
  for (int j = 0; j < 12; ++j)
#pragma unroll
    for (int i = 0; i < 8; ++i)
      ctx[(size_t)(b * Ss + q0 + i + 8 * hf) * Dd + h * HDd + j * 16 + nl] =
          __float2bfloat16(acc[j][i]);
}

// ---------------- V transpose: QKV[b,s,2D+h*HD+d] -> Vt[b,h,d,s] ----------------
__global__ void __launch_bounds__(256) build_vt(const __hip_bfloat16* __restrict__ QKV,
                                                __hip_bfloat16* __restrict__ Vt) {
  int id = blockIdx.x * 256 + threadIdx.x;           // B*NH*HD*S = 12582912
  int s = id % Ss;
  int d = (id / Ss) % HDd;
  int h = (id / (Ss * HDd)) % NHh;
  int b = id / (Ss * HDd * NHh);
  Vt[id] = QKV[(size_t)(b * Ss + s) * D3 + 2 * Dd + h * HDd + d];
}

// ---------------- mean over S of ctx (bf16) -> f32 + bf16 ----------------
__global__ void __launch_bounds__(256) mean_ctx(const __hip_bfloat16* __restrict__ ctx,
                                                float* __restrict__ mf,
                                                __hip_bfloat16* __restrict__ mb) {
  int id = blockIdx.x * 256 + threadIdx.x;           // B*D = 12288
  if (id >= Bb * Dd) return;
  int b = id / Dd, d = id % Dd;
  float s = 0.0f;
  for (int t = 0; t < Ss; ++t) s += __bfloat162float(ctx[(size_t)(b * Ss + t) * Dd + d]);
  s *= (1.0f / Ss);
  mf[id] = s;
  mb[id] = __float2bfloat16(s);
}

// ---------------- idx[b] = sum(mask[b]) - 1 ----------------
__global__ void compute_idx(const int* __restrict__ mask, int* __restrict__ idx) {
  int b = threadIdx.x;
  if (b >= Bb) return;
  int s = 0;
  for (int j = 0; j < Ss; ++j) s += mask[b * Ss + j];
  idx[b] = s - 1;
}

// ---------------- scatter se1/se2 (f32 outputs + bf16 mirrors) ----------------
__global__ void __launch_bounds__(256) scatter_se(const float* __restrict__ x,
                                                  const float* __restrict__ e1,
                                                  const float* __restrict__ e2,
                                                  const int* __restrict__ idx,
                                                  float* __restrict__ se1f, __hip_bfloat16* __restrict__ se1b,
                                                  float* __restrict__ se2f, __hip_bfloat16* __restrict__ se2b) {
  int id = blockIdx.x * 256 + threadIdx.x;           // B*S*D
  int d = id % Dd;
  int s = (id / Dd) % Ss;
  int b = id / (Dd * Ss);
  float xv = x[id];
  int ib = idx[b];
  float ev1 = e1[b * Dd + d], ev2 = e2[b * Dd + d];
  // se1: row0=e1 then row idx=e2 (e2 wins at idx); se2: row idx=e2 then row0=e1 (e1 wins at 0)
  float v1 = (s == ib) ? ev2 : ((s == 0) ? ev1 : xv);
  float v2 = (s == 0) ? ev1 : ((s == ib) ? ev2 : xv);
  se1f[id] = v1; se1b[id] = __float2bfloat16(v1);
  se2f[id] = v2; se2b[id] = __float2bfloat16(v2);
}

// ---------------- residual + LayerNorm, wave per row ----------------
__global__ void __launch_bounds__(256) ln_residual(const float* __restrict__ hin,
                                                   const float* __restrict__ ff,
                                                   const float* __restrict__ g,
                                                   const float* __restrict__ be,
                                                   float* __restrict__ hout,
                                                   __hip_bfloat16* __restrict__ hb) {
  int wave = threadIdx.x >> 5, lane = threadIdx.x & 31;
  int row = blockIdx.x * 8 + wave;                   // rows = B*S
  const float* pr = hin + (size_t)row * Dd;
  const float* pf = ff + (size_t)row * Dd;
  float s = 0.0f;
  for (int j = lane; j < Dd; j += 32) s += pr[j] + pf[j];
#pragma unroll
  for (int o = 16; o > 0; o >>= 1) s += __shfl_xor(s, o, 32);
  float mean = s * (1.0f / Dd);
  float v = 0.0f;
  for (int j = lane; j < Dd; j += 32) { float t = pr[j] + pf[j] - mean; v += t * t; }
#pragma unroll
  for (int o = 16; o > 0; o >>= 1) v += __shfl_xor(v, o, 32);
  float inv = rsqrtf(v * (1.0f / Dd) + 1e-12f);
  for (int j = lane; j < Dd; j += 32) {
    float y = (pr[j] + pf[j] - mean) * inv * g[j] + be[j];
    hout[(size_t)row * Dd + j] = y;
    hb[(size_t)row * Dd + j] = __float2bfloat16(y);
  }
}

// ---------------- small-N classifier: out[row,c] = dot(X[row], W[c]) + b[c], N<=3 ----------------
__global__ void __launch_bounds__(256) rowwise_cls(const float* __restrict__ X,
                                                   const float* __restrict__ W,
                                                   const float* __restrict__ bias,
                                                   float* __restrict__ out, int rows, int N) {
  int wave = threadIdx.x >> 5, lane = threadIdx.x & 31;
  int row = blockIdx.x * 8 + wave;
  if (row >= rows) return;
  float acc[3] = {0.0f, 0.0f, 0.0f};
  for (int j = lane; j < Dd; j += 32) {
    float xv = X[(size_t)row * Dd + j];
#pragma unroll
    for (int c = 0; c < 3; ++c)
      if (c < N) acc[c] += xv * W[c * Dd + j];
  }
#pragma unroll
  for (int c = 0; c < 3; ++c) {
    if (c >= N) break;
    float s = acc[c];
#pragma unroll
    for (int o = 16; o > 0; o >>= 1) s += __shfl_xor(s, o, 32);
    if (lane == 0) out[(size_t)row * N + c] = s + bias[c];
  }
}

// =================================================================================
extern "C" void kernel_launch(void* const* d_in, const int* in_sizes, int n_in,
                              void* d_out, int out_size, void* d_ws, size_t ws_size,
                              hipStream_t stream) {
  (void)in_sizes; (void)n_in; (void)out_size; (void)ws_size;
  const float* X        = (const float*)d_in[0];
  const int*   mask     = (const int*)d_in[1];
  const float* mha_in_w = (const float*)d_in[2];
  const float* mha_in_b = (const float*)d_in[3];
  const float* mha_out_w= (const float*)d_in[4];
  const float* mha_out_b= (const float*)d_in[5];
  const float* asp_w    = (const float*)d_in[6];
  const float* asp_b    = (const float*)d_in[7];
  const float* opi_w    = (const float*)d_in[8];
  const float* opi_b    = (const float*)d_in[9];
  const float* ia_w     = (const float*)d_in[10];
  const float* ia_b     = (const float*)d_in[11];
  const float* io_w     = (const float*)d_in[12];
  const float* io_b     = (const float*)d_in[13];
  const float* fi_w     = (const float*)d_in[14];
  const float* fi_b     = (const float*)d_in[15];
  const float* fo_w     = (const float*)d_in[16];
  const float* fo_b     = (const float*)d_in[17];
  const float* fg       = (const float*)d_in[18];
  const float* fb       = (const float*)d_in[19];
  const float* ri_w     = (const float*)d_in[20];
  const float* ri_b     = (const float*)d_in[21];
  const float* ro_w     = (const float*)d_in[22];
  const float* ro_b     = (const float*)d_in[23];
  const float* rg       = (const float*)d_in[24];
  const float* rb       = (const float*)d_in[25];
  const float* ca_w     = (const float*)d_in[26];
  const float* ca_b     = (const float*)d_in[27];
  const float* co_w     = (const float*)d_in[28];
  const float* co_b     = (const float*)d_in[29];

  float* out = (float*)d_out;
  const size_t OUT_LA = 0, OUT_LO = 49152, OUT_X = 98304;
  const size_t OUT_SE1 = 12681216, OUT_SE2 = 25264128;
  const size_t OUT_IMPA = 37847040, OUT_IMPO = 37847072;
  const size_t OUT_E1 = 37847104, OUT_E2 = 37859392;

  // ---- workspace carve-up ----
  char* w = (char*)d_ws;
  size_t off = 0;
  auto alloc = [&](size_t bytes) { size_t r = off; off = (off + bytes + 255) & ~(size_t)255; return r; };
  const size_t BSD = (size_t)Bb * Ss * Dd;
  __hip_bfloat16* XB   = (__hip_bfloat16*)(w + alloc(BSD * 2));
  __hip_bfloat16* QKVB = (__hip_bfloat16*)(w + alloc((size_t)Bb * Ss * D3 * 2));
  __hip_bfloat16* VT   = (__hip_bfloat16*)(w + alloc(BSD * 2));
  __hip_bfloat16* CTXB = (__hip_bfloat16*)(w + alloc(BSD * 2));
  __hip_bfloat16* WIN  = (__hip_bfloat16*)(w + alloc((size_t)D3 * Dd * 2));
  __hip_bfloat16* WOUT = (__hip_bfloat16*)(w + alloc((size_t)Dd * Dd * 2));
  __hip_bfloat16* WASP = (__hip_bfloat16*)(w + alloc((size_t)Dd * Dd * 2));
  __hip_bfloat16* WOPI = (__hip_bfloat16*)(w + alloc((size_t)Dd * Dd * 2));
  __hip_bfloat16* WFI  = (__hip_bfloat16*)(w + alloc((size_t)2 * DFFf * Dd * 2));
  __hip_bfloat16* WFO  = (__hip_bfloat16*)(w + alloc((size_t)2 * Dd * DFFf * 2));
  __hip_bfloat16* WRI  = (__hip_bfloat16*)(w + alloc((size_t)2 * DFFf * Dd * 2));
  __hip_bfloat16* WRO  = (__hip_bfloat16*)(w + alloc((size_t)2 * Dd * DFFf * 2));
  float*          MEANF= (float*)(w + alloc((size_t)Bb * Dd * 4));
  __hip_bfloat16* MEANB= (__hip_bfloat16*)(w + alloc((size_t)Bb * Dd * 2));
  float*          EMBF = (float*)(w + alloc((size_t)Bb * Dd * 4));
  __hip_bfloat16* EMBB = (__hip_bfloat16*)(w + alloc((size_t)Bb * Dd * 2));
  __hip_bfloat16* H1B  = (__hip_bfloat16*)(w + alloc(BSD * 2));
  __hip_bfloat16* H2B  = (__hip_bfloat16*)(w + alloc(BSD * 2));
  __hip_bfloat16* INTB = (__hip_bfloat16*)(w + alloc((size_t)Bb * Ss * DFFf * 2));
  float*          TMPF = (float*)(w + alloc(BSD * 4));
  int*            IDX  = (int*)(w + alloc(256));

  auto cvt = [&](const float* src, __hip_bfloat16* dst, int n) {
    cvt_f32_bf16<<<(n + 255) / 256, 256, 0, stream>>>(src, dst, n);
  };

  // 1) precision conversion (one-time)
  cvt(X, XB, (int)BSD);
  cvt(mha_in_w, WIN, D3 * Dd);
  cvt(mha_out_w, WOUT, Dd * Dd);
  cvt(asp_w, WASP, Dd * Dd);
  cvt(opi_w, WOPI, Dd * Dd);
  cvt(fi_w, WFI, 2 * DFFf * Dd);
  cvt(fo_w, WFO, 2 * Dd * DFFf);
  cvt(ri_w, WRI, 2 * DFFf * Dd);
  cvt(ro_w, WRO, 2 * Dd * DFFf);
  compute_idx<<<1, 16, 0, stream>>>(mask, IDX);

  const int M = Bb * Ss;   // 16384
  // 2) QKV projection: [16384,768] x [2304,768]^T -> bf16 QKV
  gemm_bf16_wmma<<<dim3(D3 / 128, M / 128), 256, 0, stream>>>(
      XB, Dd, WIN, Dd, mha_in_b, nullptr, QKVB, D3, M, Dd, 0);

  // 3) attention
  build_vt<<<(int)(BSD / 256), 256, 0, stream>>>(QKVB, VT);
  attn_wmma<<<Bb * NHh * (Ss / 16), 32, 16 * Ss * 4 + 16 * Ss * 2, stream>>>(QKVB, VT, mask, CTXB);

  // 4) embedding = mean_s(ctx) @ Wo^T + bo   (mean commutes with the linear map)
  mean_ctx<<<(Bb * Dd + 255) / 256, 256, 0, stream>>>(CTXB, MEANF, MEANB);
  gemm_bf16_wmma<<<dim3(Dd / 128, 1), 256, 0, stream>>>(
      MEANB, Dd, WOUT, Dd, mha_out_b, EMBF, EMBB, Dd, Bb, Dd, 0);

  // 5) e1 / e2 (f32 straight into d_out regions)
  gemm_bf16_wmma<<<dim3(Dd / 128, 1), 256, 0, stream>>>(
      EMBB, Dd, WASP, Dd, asp_b, out + OUT_E1, nullptr, Dd, Bb, Dd, 0);
  gemm_bf16_wmma<<<dim3(Dd / 128, 1), 256, 0, stream>>>(
      EMBB, Dd, WOPI, Dd, opi_b, out + OUT_E2, nullptr, Dd, Bb, Dd, 0);

  // 6) implicit classifiers
  rowwise_cls<<<2, 256, 0, stream>>>(out + OUT_E1, ia_w, ia_b, out + OUT_IMPA, Bb, 2);
  rowwise_cls<<<2, 256, 0, stream>>>(out + OUT_E2, io_w, io_b, out + OUT_IMPO, Bb, 2);

  // 7) scatter into se1/se2 (f32 outputs double as decoder state; bf16 mirrors for GEMMs)
  scatter_se<<<(int)(BSD / 256), 256, 0, stream>>>(
      X, out + OUT_E1, out + OUT_E2, IDX,
      out + OUT_SE1, H1B, out + OUT_SE2, H2B);

  // 8) decoder stacks: [inter=gelu(h@Wi^T+bi)] -> [out=inter@Wo^T+bo] -> LN(out+h)
  for (int st = 0; st < 2; ++st) {
    float* hf = (st == 0) ? out + OUT_SE1 : out + OUT_SE2;
    __hip_bfloat16* hb = (st == 0) ? H1B : H2B;
    const __hip_bfloat16* Wi = (st == 0) ? WFI : WRI;
    const __hip_bfloat16* Wo = (st == 0) ? WFO : WRO;
    const float* bi = (st == 0) ? fi_b : ri_b;
    const float* bo = (st == 0) ? fo_b : ro_b;
    const float* lg = (st == 0) ? fg : rg;
    const float* lb = (st == 0) ? fb : rb;
    for (int l = 0; l < 2; ++l) {
      gemm_bf16_wmma<<<dim3(DFFf / 128, M / 128), 256, 0, stream>>>(
          hb, Dd, Wi + (size_t)l * DFFf * Dd, Dd, bi + (size_t)l * DFFf,
          nullptr, INTB, DFFf, M, Dd, 1 /*gelu*/);
      gemm_bf16_wmma<<<dim3(Dd / 128, M / 128), 256, 0, stream>>>(
          INTB, DFFf, Wo + (size_t)l * Dd * DFFf, DFFf, bo + (size_t)l * Dd,
          TMPF, nullptr, Dd, M, DFFf, 0);
      ln_residual<<<M / 8, 256, 0, stream>>>(
          hf, TMPF, lg + (size_t)l * Dd, lb + (size_t)l * Dd, hf, hb);
    }
  }

  // 9) final classifiers
  rowwise_cls<<<M / 8, 256, 0, stream>>>(out + OUT_SE1, ca_w, ca_b, out + OUT_LA, M, 3);
  rowwise_cls<<<M / 8, 256, 0, stream>>>(out + OUT_SE2, co_w, co_b, out + OUT_LO, M, 3);

  // 10) x output is the untouched input
  hipMemcpyAsync(out + OUT_X, X, BSD * sizeof(float), hipMemcpyDeviceToDevice, stream);
}
